// CasualSelfAttention_19713899888819
// MI455X (gfx1250) — matrix operand order repaired and checked by reading, not compiled, not run on hardware
//
#include <hip/hip_runtime.h>
#include <hip/hip_bf16.h>
#include <stdint.h>

typedef __bf16 bf16;
typedef __attribute__((ext_vector_type(16))) __bf16 v16bf;
typedef __attribute__((ext_vector_type(8)))  float   v8f;
typedef __attribute__((ext_vector_type(4)))  unsigned int u32x4;
typedef __attribute__((ext_vector_type(4)))  int i32x4;
typedef __attribute__((ext_vector_type(8)))  int i32x8;

#if defined(__has_builtin)
#  if __has_builtin(__builtin_amdgcn_tensor_load_to_lds)
#    define USE_TDM 1
#  endif
#endif
#ifndef USE_TDM
#  define USE_TDM 0
#endif

// ---------------------------------------------------------------------------
// WMMA fragment loaders (CDNA5 ISA layouts), all 2x16B vector loads
// ---------------------------------------------------------------------------
union frag_cast { u32x4 u[2]; v16bf v; };

// A-matrix 16x32 bf16 chunk, row-major source, ld in elements.
// lanes 0-15: K = {0..7, 16..23}; lanes 16-31: K = {8..15, 24..31}.
__device__ __forceinline__ v16bf load_a_frag(const bf16* __restrict__ src, int ld, int lane) {
    const int m  = lane & 15;
    const int kb = (lane < 16) ? 0 : 8;
    const bf16* row = src + (size_t)m * ld;
    frag_cast f;
    f.u[0] = *(const u32x4*)(row + kb);       // K = kb .. kb+7
    f.u[1] = *(const u32x4*)(row + 16 + kb);  // K = 16+kb .. 16+kb+7
    return f.v;
}

// B-matrix 32x16 where source is stored N-major: element (k,n) = src[n*ld + k].
// lanes 0-15: col n=l, K={0..7,16..23}; lanes 16-31: col n=l-16, K={8..15,24..31}.
__device__ __forceinline__ v16bf load_bt_frag(const bf16* __restrict__ src, int ld, int lane) {
    const int n  = lane & 15;
    const int kb = (lane < 16) ? 0 : 8;
    const bf16* row = src + (size_t)n * ld + kb;
    frag_cast f;
    f.u[0] = *(const u32x4*)(row);
    f.u[1] = *(const u32x4*)(row + 16);
    return f.v;
}

// ---------------------------------------------------------------------------
// TDM: build D# per CDNA5 ISA and issue tensor_load_to_lds.
// 4-byte elements. tile_w4/pitch4/width4 in dwords; tile_h rows (1 => linear).
// Toolchain uses the 6-arg builtin: (g0, g1, g2, g3, g4, cpol).
// ---------------------------------------------------------------------------
#if USE_TDM
__device__ __forceinline__ void tdm_load(unsigned lds_addr, const void* g,
                                         unsigned tile_w4, unsigned tile_h,
                                         unsigned pitch4, unsigned width4) {
    unsigned long long ga = (unsigned long long)(uintptr_t)g;
    u32x4 g0 = { 1u,                                   // count=1, valid
                 lds_addr,                             // lds_addr [63:32]
                 (unsigned)ga,                         // global_addr low
                 (unsigned)(ga >> 32) | 0x80000000u }; // addr high | type=2
    i32x8 g1;
    g1[0] = (int)(2u << 16);                 // data_size = 4 bytes
    g1[1] = (int)(width4 << 16);             // tensor_dim0[15:0]
    g1[2] = (int)((width4 >> 16) | (tile_h << 16)); // dim0 hi | tensor_dim1 lo
    g1[3] = (int)(tile_w4 << 16);            // tile_dim0
    g1[4] = (int)tile_h;                     // tile_dim1 (tile_dim2 = 0)
    g1[5] = (int)pitch4;                     // tensor_dim0_stride low32
    g1[6] = 0;
    g1[7] = 0;
    i32x4 z4 = { 0, 0, 0, 0 };
    i32x8 z8 = { 0, 0, 0, 0, 0, 0, 0, 0 };
    __builtin_amdgcn_tensor_load_to_lds(g0, g1, z4, z4, z8, 0);
    asm volatile("" ::: "memory");           // LDS is written behind compiler's back
}
#endif

// ---------------------------------------------------------------------------
// fp32 -> bf16 conversion
// ---------------------------------------------------------------------------
__global__ void cvt_f32_bf16(const float* __restrict__ in, bf16* __restrict__ out, int n) {
    int i = blockIdx.x * blockDim.x + threadIdx.x;
    if (i < n) out[i] = (bf16)in[i];
}

// ---------------------------------------------------------------------------
// Shared tile compute: 256 threads = 8 waves, block tile 128(M) x 128(N),
// K-step 32. Waves: 4 along M x 2 along N; each wave: 2 Mx16 x 4 Nx16 tiles.
// B tile staged transposed: Bt[n][k], ld = 32.
// ---------------------------------------------------------------------------
__device__ __forceinline__ void mma_step(const bf16* __restrict__ As,
                                         const bf16* __restrict__ Bt,
                                         int wm, int wn, int lane, v8f acc[2][4]) {
    v16bf a[2];
    a[0] = load_a_frag(As + (size_t)(wm * 32) * 32, 32, lane);
    a[1] = load_a_frag(As + (size_t)(wm * 32 + 16) * 32, 32, lane);
#pragma unroll
    for (int nt = 0; nt < 4; ++nt) {
        v16bf b = load_bt_frag(Bt + (size_t)(wn * 64 + nt * 16) * 32, 32, lane);
        acc[0][nt] = __builtin_amdgcn_wmma_f32_16x16x32_bf16(
            false, a[0], false, b, (short)0, acc[0][nt], false, false);
        acc[1][nt] = __builtin_amdgcn_wmma_f32_16x16x32_bf16(
            false, a[1], false, b, (short)0, acc[1][nt], false, false);
    }
}

// Stage A row-major [128][32] and B transposed [128][32] for one K-step.
__device__ __forceinline__ void stage_tiles(const bf16* __restrict__ A, int lda,
                                            const bf16* __restrict__ Bw, int ldb,
                                            bf16* __restrict__ As, bf16* __restrict__ Bt,
                                            int bm, int bn, int k0, int K, int tid) {
#pragma unroll
    for (int i = tid; i < 512; i += 256) {
        int row = i >> 2, c4 = i & 3;
        const u32x4* src = (const u32x4*)(A + (size_t)(bm * 128 + row) * lda + k0) + c4;
        ((u32x4*)As)[i] = *src;
        if (k0 + 32 < K) __builtin_prefetch((const char*)src + 64, 0, 3);
    }
#pragma unroll
    for (int i = tid; i < 512; i += 256) {
        int krow = i >> 4, c4 = i & 15;
        u32x4 d = *((const u32x4*)(Bw + (size_t)(k0 + krow) * ldb + bn * 128) + c4);
        const bf16* e = (const bf16*)&d;
        int n0 = c4 * 8;
#pragma unroll
        for (int j = 0; j < 8; ++j)
            Bt[(size_t)(n0 + j) * 32 + krow] = e[j];
    }
}

// ---------------------------------------------------------------------------
// QKV GEMM: [8192,1024] x [1024,3072] + bias.
// q,k: [B,H,T,64] bf16;  v stored TRANSPOSED: [B,H,64,T] bf16.
// ---------------------------------------------------------------------------
__global__ void __launch_bounds__(256)
gemm_qkv_kernel(const bf16* __restrict__ A, const bf16* __restrict__ Bw,
                const float* __restrict__ bias,
                bf16* __restrict__ q, bf16* __restrict__ k, bf16* __restrict__ v) {
    constexpr int K = 1024, N = 3072, T = 2048;
    __shared__ __align__(16) bf16 As[128 * 32];
    __shared__ __align__(16) bf16 Bt[128 * 32];
    const int tid = threadIdx.x, lane = tid & 31;
    const int w = tid >> 5, wm = w >> 1, wn = w & 1;
    const int bm = blockIdx.x, bn = blockIdx.y;
    v8f acc[2][4] = {};

    for (int k0 = 0; k0 < K; k0 += 32) {
        stage_tiles(A, K, Bw, N, As, Bt, bm, bn, k0, K, tid);
        __syncthreads();
        mma_step(As, Bt, wm, wn, lane, acc);
        __syncthreads();
    }

    const int moff = (lane >> 4) << 3;
    const int nl = lane & 15;
#pragma unroll
    for (int mt = 0; mt < 2; ++mt)
#pragma unroll
        for (int nt = 0; nt < 4; ++nt)
#pragma unroll
            for (int r = 0; r < 8; ++r) {
                int gm = bm * 128 + wm * 32 + mt * 16 + moff + r;
                int gn = bn * 128 + wn * 64 + nt * 16 + nl;
                float val = acc[mt][nt][r] + bias[gn];
                int sec = gn >> 10, cc = gn & 1023;
                int h = cc >> 6, dd = cc & 63;
                int b = gm >> 11, t = gm & 2047;
                int bh = b * 16 + h;
                if (sec == 0)
                    q[((size_t)bh * T + t) * 64 + dd] = (bf16)val;
                else if (sec == 1)
                    k[((size_t)bh * T + t) * 64 + dd] = (bf16)val;
                else
                    v[((size_t)bh * 64 + dd) * T + t] = (bf16)val;   // V^T
            }
}

// ---------------------------------------------------------------------------
// Flash attention: block = 64 query rows of one (b,h). 128 threads = 4 waves.
// K tile staged row-major [key][d]; V^T tile staged [d][key]; both give
// contiguous B-fragment reads. K/V staged by the Tensor Data Mover.
// ---------------------------------------------------------------------------
__global__ void __launch_bounds__(128)
attn_kernel(const bf16* __restrict__ q, const bf16* __restrict__ kk,
            const bf16* __restrict__ vt, bf16* __restrict__ y) {
    constexpr int T = 2048, D = 64;
    __shared__ __align__(16) bf16 kbuf[64 * 64];   // [key][d]
    __shared__ __align__(16) bf16 vbuf[64 * 64];   // [d][key]
    __shared__ __align__(16) bf16 pbuf[4][16 * 64];
    const int tid = threadIdx.x, lane = tid & 31, w = tid >> 5;
    const int bh = blockIdx.x, qblk = blockIdx.y;
    const size_t headbase = (size_t)bh * T * D;

    const bf16* qbase = q + headbase + (size_t)(qblk * 64 + w * 16) * D;
    v16bf qf[2];
    qf[0] = load_a_frag(qbase, D, lane);
    qf[1] = load_a_frag(qbase + 32, D, lane);

    v8f O[4] = {};
    float mrow[8], lrow[8];
#pragma unroll
    for (int r = 0; r < 8; ++r) { mrow[r] = -3.0e38f; lrow[r] = 0.0f; }

    const int moff = (lane >> 4) << 3;
    const int nl = lane & 15;
    const int gq0 = qblk * 64 + w * 16 + moff;

    for (int kt = 0; kt <= qblk; ++kt) {
        __syncthreads();
#if USE_TDM
        if (w == 0) {
            // K tile: 64 rows x 64 bf16 fully contiguous => linear 2048 dwords
            tdm_load((unsigned)(uintptr_t)kbuf,
                     kk + headbase + (size_t)kt * 64 * D,
                     2048u, 1u, 2048u, 65536u);
            __builtin_amdgcn_s_wait_tensorcnt(0);
        } else if (w == 1) {
            // V^T tile: 64 rows x 32 dwords, pitch T/2 = 1024 dwords
            tdm_load((unsigned)(uintptr_t)vbuf,
                     vt + (size_t)bh * D * T + (size_t)kt * 64,
                     32u, 64u, 1024u, 1024u);
            __builtin_amdgcn_s_wait_tensorcnt(0);
        }
#else
        for (int i = tid; i < 512; i += 128)
            ((u32x4*)kbuf)[i] = ((const u32x4*)(kk + headbase + (size_t)kt * 64 * D))[i];
        for (int i = tid; i < 512; i += 128) {
            int row = i >> 3, c4 = i & 7;
            ((u32x4*)vbuf)[i] =
                *((const u32x4*)(vt + (size_t)bh * D * T + (size_t)row * T + kt * 64) + c4);
        }
#endif
        __syncthreads();

        // S = Q * K^T
        v8f s[4];
#pragma unroll
        for (int nt = 0; nt < 4; ++nt) {
            v8f a = {};
#pragma unroll
            for (int c = 0; c < 2; ++c) {
                v16bf b = load_bt_frag(kbuf + (size_t)(nt * 16) * D + c * 32, D, lane);
                a = __builtin_amdgcn_wmma_f32_16x16x32_bf16(
                    false, qf[c], false, b, (short)0, a, false, false);
            }
            s[nt] = a;
        }

        // scale + causal mask
        const int gk0 = kt * 64 + nl;
#pragma unroll
        for (int nt = 0; nt < 4; ++nt)
#pragma unroll
            for (int r = 0; r < 8; ++r) {
                float sv = s[nt][r] * 0.125f;  // 1/sqrt(64)
                if (gk0 + nt * 16 > gq0 + r) sv = -3.0e38f;
                s[nt][r] = sv;
            }

        // online softmax: C-layout rows live across 16-lane halves
#pragma unroll
        for (int r = 0; r < 8; ++r) {
            float m_new = mrow[r];
#pragma unroll
            for (int nt = 0; nt < 4; ++nt) m_new = fmaxf(m_new, s[nt][r]);
#pragma unroll
            for (int off = 1; off < 16; off <<= 1)
                m_new = fmaxf(m_new, __shfl_xor(m_new, off, 32));
            float alpha = __expf(mrow[r] - m_new);
            float rsum = 0.0f;
#pragma unroll
            for (int nt = 0; nt < 4; ++nt) {
                float p = __expf(s[nt][r] - m_new);
                s[nt][r] = p;
                rsum += p;
            }
#pragma unroll
            for (int off = 1; off < 16; off <<= 1)
                rsum += __shfl_xor(rsum, off, 32);
            lrow[r] = lrow[r] * alpha + rsum;
            mrow[r] = m_new;
#pragma unroll
            for (int nt = 0; nt < 4; ++nt) O[nt][r] *= alpha;
        }

        // re-stage P through LDS into A-layout, then O += P * V
        bf16* pb = &pbuf[w][0];
#pragma unroll
        for (int nt = 0; nt < 4; ++nt)
#pragma unroll
            for (int r = 0; r < 8; ++r)
                pb[(moff + r) * 64 + nt * 16 + nl] = (bf16)s[nt][r];
        __syncthreads();
#pragma unroll
        for (int c = 0; c < 2; ++c) {
            v16bf a = load_a_frag(pb + c * 32, 64, lane);
#pragma unroll
            for (int nd = 0; nd < 4; ++nd) {
                v16bf b = load_bt_frag(vbuf + (size_t)(nd * 16) * D + c * 32, D, lane);
                O[nd] = __builtin_amdgcn_wmma_f32_16x16x32_bf16(
                    false, a, false, b, (short)0, O[nd], false, false);
            }
        }
    }

    // epilogue: normalize, write y as [B, T, H*64] (row-major [8192,1024])
    const int b = bh >> 4, h = bh & 15;
#pragma unroll
    for (int nt = 0; nt < 4; ++nt)
#pragma unroll
        for (int r = 0; r < 8; ++r) {
            float val = O[nt][r] / lrow[r];
            int t = gq0 + r;
            y[((size_t)b * T + t) * 1024 + h * 64 + nt * 16 + nl] = (bf16)val;
        }
}

// ---------------------------------------------------------------------------
// Projection GEMM: [8192,1024] x [1024,1024] + bias -> fp32 out
// ---------------------------------------------------------------------------
__global__ void __launch_bounds__(256)
gemm_proj_kernel(const bf16* __restrict__ A, const bf16* __restrict__ Bw,
                 const float* __restrict__ bias, float* __restrict__ out) {
    constexpr int K = 1024, N = 1024;
    __shared__ __align__(16) bf16 As[128 * 32];
    __shared__ __align__(16) bf16 Bt[128 * 32];
    const int tid = threadIdx.x, lane = tid & 31;
    const int w = tid >> 5, wm = w >> 1, wn = w & 1;
    const int bm = blockIdx.x, bn = blockIdx.y;
    v8f acc[2][4] = {};

    for (int k0 = 0; k0 < K; k0 += 32) {
        stage_tiles(A, K, Bw, N, As, Bt, bm, bn, k0, K, tid);
        __syncthreads();
        mma_step(As, Bt, wm, wn, lane, acc);
        __syncthreads();
    }

    const int moff = (lane >> 4) << 3;
    const int nl = lane & 15;
#pragma unroll
    for (int mt = 0; mt < 2; ++mt)
#pragma unroll
        for (int nt = 0; nt < 4; ++nt)
#pragma unroll
            for (int r = 0; r < 8; ++r) {
                int gm = bm * 128 + wm * 32 + mt * 16 + moff + r;
                int gn = bn * 128 + wn * 64 + nt * 16 + nl;
                out[(size_t)gm * N + gn] = acc[mt][nt][r] + bias[gn];
            }
}

// ---------------------------------------------------------------------------
// Host launch
// ---------------------------------------------------------------------------
extern "C" void kernel_launch(void* const* d_in, const int* in_sizes, int n_in,
                              void* d_out, int out_size, void* d_ws, size_t ws_size,
                              hipStream_t stream) {
    const float* x      = (const float*)d_in[0];
    const float* attn_w = (const float*)d_in[1];
    const float* attn_b = (const float*)d_in[2];
    const float* proj_w = (const float*)d_in[3];
    const float* proj_b = (const float*)d_in[4];
    float* out = (float*)d_out;

    const size_t BT = 4 * 2048, C = 1024;
    bf16* xb     = (bf16*)d_ws;          // [8192,1024]   (reused as y later)
    bf16* wqkvb  = xb + BT * C;          // [1024,3072]
    bf16* wprojb = wqkvb + C * 3 * C;    // [1024,1024]
    bf16* qb     = wprojb + C * C;       // [B,H,T,64]
    bf16* kb     = qb + BT * C;          // [B,H,T,64]
    bf16* vb     = kb + BT * C;          // [B,H,64,T]  (transposed)
    bf16* yb     = xb;                   // alias: x dead after QKV GEMM

    int n1 = (int)(BT * C);
    cvt_f32_bf16<<<(n1 + 255) / 256, 256, 0, stream>>>(x, xb, n1);
    int n2 = (int)(C * 3 * C);
    cvt_f32_bf16<<<(n2 + 255) / 256, 256, 0, stream>>>(attn_w, wqkvb, n2);
    int n3 = (int)(C * C);
    cvt_f32_bf16<<<(n3 + 255) / 256, 256, 0, stream>>>(proj_w, wprojb, n3);

    gemm_qkv_kernel<<<dim3(64, 24), 256, 0, stream>>>(xb, wqkvb, attn_b, qb, kb, vb);
    attn_kernel<<<dim3(64, 32), 128, 0, stream>>>(qb, kb, vb, yb);
    gemm_proj_kernel<<<dim3(64, 8), 256, 0, stream>>>(yb, wprojb, proj_b, out);
}
